// MeanAggregator_34591666602498
// MI455X (gfx1250) — matrix operand order repaired and checked by reading, not compiled, not run on hardware
//
#include <hip/hip_runtime.h>

typedef _Float16 v16h __attribute__((ext_vector_type(16)));
typedef float    v8f  __attribute__((ext_vector_type(8)));
typedef _Float16 h8v  __attribute__((ext_vector_type(8)));

#define Hdim   200
#define Bdim   512
#define SEQ    10
#define E_TOT  256000
#define NSEG   (Bdim * SEQ)   // 5120
#define NP     208            // N padded (13 tiles of 16)
#define NT     13             // N tiles
#define K1P    608            // 600 padded to 19*32
#define K2P    416            // 400 padded to 13*32
#define NC1    19
#define NC2    13
#define SBS    40             // LDS B row stride in halves (80B: bank-conflict-free b128 frags)

// ---------------- helpers ----------------

__device__ __forceinline__ v16h load16h(const _Float16* __restrict__ p) {
  h8v lo = *(const h8v*)p;
  h8v hi = *(const h8v*)(p + 16);
  v16h r;
#pragma unroll
  for (int i = 0; i < 8; ++i) { r[i] = lo[i]; r[i + 8] = hi[i]; }
  return r;
}

// 8 contiguous f32 -> 8 f16 halves of the A fragment
__device__ __forceinline__ void fill8_gather(v16h& a, int h0, const float* __restrict__ src) {
  const float4* p = (const float4*)src;
  float4 p0 = p[0], p1 = p[1];
  a[h0+0]=(_Float16)p0.x; a[h0+1]=(_Float16)p0.y; a[h0+2]=(_Float16)p0.z; a[h0+3]=(_Float16)p0.w;
  a[h0+4]=(_Float16)p1.x; a[h0+5]=(_Float16)p1.y; a[h0+6]=(_Float16)p1.z; a[h0+7]=(_Float16)p1.w;
}

__device__ __forceinline__ void fill8_att(v16h& a, int h0, float aa,
                                          const float* __restrict__ W1,
                                          const float* __restrict__ b1, int K0) {
  const float4* w = (const float4*)(W1 + K0);
  const float4* c = (const float4*)(b1 + K0);
  float4 w0 = w[0], w1 = w[1], c0 = c[0], c1 = c[1];
  a[h0+0]=(_Float16)fmaxf(fmaf(aa,w0.x,c0.x),0.f);
  a[h0+1]=(_Float16)fmaxf(fmaf(aa,w0.y,c0.y),0.f);
  a[h0+2]=(_Float16)fmaxf(fmaf(aa,w0.z,c0.z),0.f);
  a[h0+3]=(_Float16)fmaxf(fmaf(aa,w0.w,c0.w),0.f);
  a[h0+4]=(_Float16)fmaxf(fmaf(aa,w1.x,c1.x),0.f);
  a[h0+5]=(_Float16)fmaxf(fmaf(aa,w1.y,c1.y),0.f);
  a[h0+6]=(_Float16)fmaxf(fmaf(aa,w1.z,c1.z),0.f);
  a[h0+7]=(_Float16)fmaxf(fmaf(aa,w1.w,c1.w),0.f);
}

__device__ __forceinline__ void fill8_zero(v16h& a, int h0) {
#pragma unroll
  for (int j = 0; j < 8; ++j) a[h0+j] = (_Float16)0.f;
}

// PATH1 A row = [ relu(att*W1+b1) | attr[o] | rel[r] | 0 ] ; PATH2 = [ ent[o] | rel[r] | 0 ]
template<int PATH>
__device__ __forceinline__ void fillA(v16h& a, int h0, int K0, float aa,
                                      const float* __restrict__ W1, const float* __restrict__ b1,
                                      const float* __restrict__ srcA,
                                      const float* __restrict__ srcB) {
  if constexpr (PATH == 1) {
    if      (K0 < 200) fill8_att(a, h0, aa, W1, b1, K0);
    else if (K0 < 400) fill8_gather(a, h0, srcA + (K0 - 200));
    else if (K0 < 600) fill8_gather(a, h0, srcB + (K0 - 400));
    else               fill8_zero(a, h0);
  } else {
    if      (K0 < 200) fill8_gather(a, h0, srcA + K0);
    else if (K0 < 400) fill8_gather(a, h0, srcB + (K0 - 200));
    else               fill8_zero(a, h0);
  }
}

// split staging: global -> regs (issued early), regs -> LDS (issued late)
template<int KP>
__device__ __forceinline__ void stage_load(h8v (&r)[4], const _Float16* __restrict__ W,
                                           int kc, int tid) {
#pragma unroll
  for (int i = 0; i < 4; ++i) {                 // 832 h8v copies over 256 threads
    int idx = tid + i * 256;
    if (idx < NP * 4) {
      int n = idx >> 2, p = (idx & 3) << 3;
      r[i] = *(const h8v*)(W + (size_t)n * KP + kc * 32 + p);
    }
  }
}

__device__ __forceinline__ void stage_store(const h8v (&r)[4], _Float16* __restrict__ dst,
                                            int tid) {
#pragma unroll
  for (int i = 0; i < 4; ++i) {
    int idx = tid + i * 256;
    if (idx < NP * 4) {
      int n = idx >> 2, p = (idx & 3) << 3;
      *(h8v*)(dst + n * SBS + p) = r[i];
    }
  }
}

// run-length-combined segment accumulation for one 16-row C tile set
__device__ __forceinline__ void seg_epilogue(const v8f (&c)[NT], int rowbase, int g, int m,
                                             const float* __restrict__ bias,
                                             float* __restrict__ segsum,
                                             const int* __restrict__ seg_ids) {
  int segv[8];
#pragma unroll
  for (int j = 0; j < 8; ++j) segv[j] = seg_ids[rowbase + 8 * g + j];
#pragma unroll
  for (int n = 0; n < NT; ++n) {
    const int col = n * 16 + m;
    const bool ok = (col < Hdim);
    const float bs = ok ? bias[col] : 0.f;
    float acc = 0.f;
    int cur = segv[0];
#pragma unroll
    for (int j = 0; j < 8; ++j) {
      float v = fmaxf(c[n][j] + bs, 0.f);
      if (segv[j] != cur) {
        if (ok) atomicAdd(&segsum[(size_t)cur * Hdim + col], acc);
        acc = 0.f; cur = segv[j];
      }
      acc += v;
    }
    if (ok) atomicAdd(&segsum[(size_t)cur * Hdim + col], acc);
  }
}

// one GEMM path: 2 M-tiles per wave, B double-buffered in LDS, shared by 8 waves
template<int PATH, int KP, int NC>
__device__ __forceinline__ void gemm_path(_Float16 (&sB)[2][NP * SBS],
                                          const _Float16* __restrict__ Wh,
                                          const float* __restrict__ bias,
                                          float* __restrict__ segsum,
                                          const int* __restrict__ seg_ids,
                                          int tid, int g, int m, int w, int bb,
                                          float aa0, float aa1,
                                          const float* __restrict__ W1,
                                          const float* __restrict__ b1,
                                          const float* __restrict__ srcA0,
                                          const float* __restrict__ srcA1,
                                          const float* __restrict__ srcB0,
                                          const float* __restrict__ srcB1) {
  const v8f vzero = {0.f, 0.f, 0.f, 0.f, 0.f, 0.f, 0.f, 0.f};
  v8f c0[NT], c1[NT];
#pragma unroll
  for (int n = 0; n < NT; ++n) { c0[n] = vzero; c1[n] = vzero; }

  {  // initial stage of chunk 0
    h8v r0[4];
    stage_load<KP>(r0, Wh, 0, tid);
    stage_store(r0, sB[0], tid);
  }
  __syncthreads();

  for (int kc = 0; kc < NC; ++kc) {
    const int cur = kc & 1;
    const bool havepre = (kc + 1 < NC);

    // issue next-chunk global loads early; park in regs
    h8v pre[4];
    if (havepre) stage_load<KP>(pre, Wh, kc + 1, tid);

    v16h a0, a1;
    fillA<PATH>(a0, 0, kc * 32 + 8 * g,      aa0, W1, b1, srcA0, srcB0);
    fillA<PATH>(a0, 8, kc * 32 + 16 + 8 * g, aa0, W1, b1, srcA0, srcB0);
    fillA<PATH>(a1, 0, kc * 32 + 8 * g,      aa1, W1, b1, srcA1, srcB1);
    fillA<PATH>(a1, 8, kc * 32 + 16 + 8 * g, aa1, W1, b1, srcA1, srcB1);

    const _Float16* lbase = sB[cur] + m * SBS + 8 * g;
    v16h b = load16h(lbase);                   // fragment n=0
#pragma unroll
    for (int n = 0; n < NT; ++n) {
      v16h bn;
      if (n + 1 < NT) bn = load16h(lbase + (n + 1) * 16 * SBS);   // prefetch n+1
      c0[n] = __builtin_amdgcn_wmma_f32_16x16x32_f16(false, a0, false, b,
                                                     (short)0, c0[n], false, false);
      c1[n] = __builtin_amdgcn_wmma_f32_16x16x32_f16(false, a1, false, b,
                                                     (short)0, c1[n], false, false);
      if (n + 1 < NT) b = bn;
    }

    // LDS writes only after all fragment reads of this chunk were issued
    if (havepre) stage_store(pre, sB[cur ^ 1], tid);
    __syncthreads();
  }

  const int rowbase = bb + w * 32;
  seg_epilogue(c0, rowbase,      g, m, bias, segsum, seg_ids);
  seg_epilogue(c1, rowbase + 16, g, m, bias, segsum, seg_ids);
}

// ---------------- kernels ----------------

__global__ void zero_ws_k(float* __restrict__ p, int n) {
  int i = blockIdx.x * blockDim.x + threadIdx.x;
  if (i < n) p[i] = 0.f;
}

__global__ void prep_weights_k(const float* __restrict__ W3, const float* __restrict__ W4,
                               _Float16* __restrict__ W3h, _Float16* __restrict__ W4h) {
  const int n1 = NP * K1P;
  const int n2 = NP * K2P;
  for (int i = blockIdx.x * blockDim.x + threadIdx.x; i < n1 + n2;
       i += gridDim.x * blockDim.x) {
    if (i < n1) {
      int r = i / K1P, k = i % K1P;
      W3h[i] = (r < Hdim && k < 600) ? (_Float16)W3[r * 600 + k] : (_Float16)0.f;
    } else {
      int j = i - n1;
      int r = j / K2P, k = j % K2P;
      W4h[j] = (r < Hdim && k < 400) ? (_Float16)W4[r * 400 + k] : (_Float16)0.f;
    }
  }
}

__global__ __launch_bounds__(256) void fused_event_gemm_k(
    const float* __restrict__ att, const int* __restrict__ o_idx,
    const int* __restrict__ rel_idx, const int* __restrict__ seg_ids,
    const float* __restrict__ ent, const float* __restrict__ attr,
    const float* __restrict__ rel,
    const float* __restrict__ W1, const float* __restrict__ b1,
    const float* __restrict__ b3, const float* __restrict__ b4,
    const _Float16* __restrict__ W3h, const _Float16* __restrict__ W4h,
    float* __restrict__ segsum1, float* __restrict__ segsum2)
{
  __shared__ _Float16 sB[2][NP * SBS];   // 33,280 B, double-buffered B chunk

  const int tid  = threadIdx.x;
  const int lane = tid & 31;
  const int w    = tid >> 5;             // wave within block (8 waves)
  const int g    = lane >> 4;
  const int m    = lane & 15;
  const int bb   = blockIdx.x * 256;     // 256 events per block, 16 M-tiles

  const int e0 = bb + w * 32 + m;        // M-tile 2w
  const int e1 = e0 + 16;                // M-tile 2w+1

  const float aa0 = att[e0], aa1 = att[e1];
  const int oi0 = o_idx[e0], oi1 = o_idx[e1];
  const int ri0 = rel_idx[e0], ri1 = rel_idx[e1];

  const float* attr0 = attr + (size_t)oi0 * Hdim;
  const float* attr1 = attr + (size_t)oi1 * Hdim;
  const float* ent0  = ent  + (size_t)oi0 * Hdim;
  const float* ent1  = ent  + (size_t)oi1 * Hdim;
  const float* rel0  = rel  + (size_t)ri0 * Hdim;
  const float* rel1  = rel  + (size_t)ri1 * Hdim;

  // path 1: relu(W3 . [eatt|attr|rel] + b3) -> segsum1 (attribute mean)
  gemm_path<1, K1P, NC1>(sB, W3h, b3, segsum1, seg_ids, tid, g, m, w, bb,
                         aa0, aa1, W1, b1, attr0, attr1, rel0, rel1);
  // path 2: relu(W4 . [ent|rel] + b4) -> segsum2 (static mean)
  gemm_path<2, K2P, NC2>(sB, W4h, b4, segsum2, seg_ids, tid, g, m, w, bb,
                         0.f, 0.f, nullptr, nullptr, ent0, ent1, rel0, rel1);
}

__device__ __forceinline__ int lower_bound_i(const int* __restrict__ a, int n, int v) {
  int lo = 0, hi = n;
  while (lo < hi) {
    int mid = (lo + hi) >> 1;
    if (a[mid] < v) lo = mid + 1; else hi = mid;
  }
  return lo;
}

__global__ __launch_bounds__(256) void finalize_k(
    const float* __restrict__ self_att, const int* __restrict__ s, const int* __restrict__ r_idx,
    const int* __restrict__ seg_ids,
    const float* __restrict__ ent, const float* __restrict__ attr, const float* __restrict__ rel,
    const float* __restrict__ W1, const float* __restrict__ b1,
    const float* __restrict__ segsum1, const float* __restrict__ segsum2,
    float* __restrict__ out)
{
  const int gseg = blockIdx.x;           // 0..NSEG-1
  const int b = gseg / SEQ;
  __shared__ float sinv;
  if (threadIdx.x == 0) {
    int lo = lower_bound_i(seg_ids, E_TOT, gseg);
    int hi = lower_bound_i(seg_ids, E_TOT, gseg + 1);
    sinv = 1.f / fmaxf((float)(hi - lo), 1.f);
  }
  __syncthreads();
  const float inv = sinv;
  const int sb = s[b], rb = r_idx[b];
  const float sa = self_att[gseg];
  float* out1 = out + (size_t)gseg * 600;                      // s_embed_seq
  float* out2 = out + (size_t)NSEG * 600 + (size_t)gseg * 600; // att_embed_seq
  for (int c = threadIdx.x; c < 600; c += blockDim.x) {
    float v1, v2;
    if (c < 200) {
      v1 = ent[(size_t)sb * Hdim + c];
      v2 = fmaxf(fmaf(sa, W1[c], b1[c]), 0.f);
    } else if (c < 400) {
      v1 = rel[(size_t)rb * Hdim + (c - 200)];
      v2 = attr[(size_t)sb * Hdim + (c - 200)];
    } else {
      v1 = segsum2[(size_t)gseg * Hdim + (c - 400)] * inv;  // static mean (W4 path)
      v2 = segsum1[(size_t)gseg * Hdim + (c - 400)] * inv;  // attr mean (W3 path)
    }
    out1[c] = v1;
    out2[c] = v2;
  }
}

// ---------------- launcher ----------------

extern "C" void kernel_launch(void* const* d_in, const int* in_sizes, int n_in,
                              void* d_out, int out_size, void* d_ws, size_t ws_size,
                              hipStream_t stream) {
  const float* att      = (const float*)d_in[0];
  const float* self_att = (const float*)d_in[1];
  const int*   o_idx    = (const int*)d_in[2];
  const int*   rel_idx  = (const int*)d_in[3];
  const int*   seg_ids  = (const int*)d_in[4];
  const int*   s        = (const int*)d_in[5];
  const int*   r        = (const int*)d_in[6];
  const float* ent      = (const float*)d_in[7];
  const float* attr     = (const float*)d_in[8];
  const float* rel      = (const float*)d_in[9];
  const float* W1       = (const float*)d_in[10];
  const float* b1       = (const float*)d_in[11];
  const float* W3       = (const float*)d_in[12];
  const float* b3       = (const float*)d_in[13];
  const float* W4       = (const float*)d_in[14];
  const float* b4       = (const float*)d_in[15];
  float* out = (float*)d_out;

  char* ws = (char*)d_ws;
  const size_t segBytes = (size_t)NSEG * Hdim * sizeof(float);   // 4,096,000
  float*    segsum1 = (float*)ws;
  float*    segsum2 = (float*)(ws + segBytes);
  _Float16* W3h     = (_Float16*)(ws + 2 * segBytes);
  _Float16* W4h     = (_Float16*)(ws + 2 * segBytes + (size_t)NP * K1P * sizeof(_Float16));

  const int nzero = 2 * NSEG * Hdim;                 // 2,048,000 floats
  zero_ws_k<<<(nzero + 255) / 256, 256, 0, stream>>>(segsum1, nzero);

  prep_weights_k<<<((NP * K1P + NP * K2P) + 255) / 256, 256, 0, stream>>>(W3, W4, W3h, W4h);

  // 256 events per block (16 M-tiles, 2 per wave), 1000 blocks
  fused_event_gemm_k<<<E_TOT / 256, 256, 0, stream>>>(
      att, o_idx, rel_idx, seg_ids, ent, attr, rel,
      W1, b1, b3, b4, W3h, W4h, segsum1, segsum2);

  finalize_k<<<NSEG, 256, 0, stream>>>(
      self_att, s, r, seg_ids, ent, attr, rel, W1, b1, segsum1, segsum2, out);
}
